// SelfAttentionModule_37245956391527
// MI455X (gfx1250) — matrix operand order repaired and checked by reading, not compiled, not run on hardware
//
#include <hip/hip_runtime.h>
#include <hip/hip_bf16.h>

typedef _Float16 v16h __attribute__((ext_vector_type(16)));
typedef float    v8f  __attribute__((ext_vector_type(8)));
typedef _Float16 v8h  __attribute__((ext_vector_type(8)));
typedef _Float16 h2   __attribute__((ext_vector_type(2)));

#define D_  512
#define H_  8
#define HD_ 64
#define B_  8
#define L_  384

static __device__ __forceinline__ v8f wmma_f16(v16h a, v16h b, v8f c) {
    // D = A(16x32 f16) * B(32x16 f16) + C(16x16 f32)
    return __builtin_amdgcn_wmma_f32_16x16x32_f16(false, a, false, b, (short)0, c, false, false);
}

// Fragment container: 16 halfs = 2 x 16B chunks
union frag16 { v16h v; v8h h[2]; };

// ---------------- async global -> LDS staging (CDNA5 path) ----------------
#if __has_builtin(__builtin_amdgcn_global_load_async_to_lds_b128)
#define ASYNC_LDS 1
typedef int v4i_ll __attribute__((vector_size(16)));                 // matches builtin pointee
typedef __attribute__((address_space(1))) v4i_ll* as1_v4i;           // global (prints as __device__)
typedef __attribute__((address_space(3))) v4i_ll* as3_v4i;           // LDS    (prints as __shared__)
#endif

static __device__ __forceinline__ void copy16_g2lds(_Float16* lds, const _Float16* g) {
#if defined(ASYNC_LDS)
    __builtin_amdgcn_global_load_async_to_lds_b128((as1_v4i)(void*)g,
                                                   (as3_v4i)(void*)lds, 0, 0);
#else
    *(v8h*)lds = *(const v8h*)g;
#endif
}

static __device__ __forceinline__ void async_join() {
#if defined(ASYNC_LDS)
#if __has_builtin(__builtin_amdgcn_s_wait_asynccnt)
    __builtin_amdgcn_s_wait_asynccnt(0);
#else
    asm volatile("s_wait_asynccnt 0x0" ::: "memory");
#endif
#endif
}

// ---------------- f32 -> f16 convert ----------------
__global__ void k_f32_to_f16(const float* __restrict__ in, _Float16* __restrict__ out, int n) {
    int i = blockIdx.x * blockDim.x + threadIdx.x;
    if (i < n) out[i] = (_Float16)in[i];
}

// ---------------- LayerNorm + cast to f16 ----------------
__global__ void k_layernorm(const float* __restrict__ x, const float* __restrict__ g,
                            const float* __restrict__ bta, _Float16* __restrict__ xh) {
    __shared__ float red[256];
    const int tid = threadIdx.x;
    const size_t base = (size_t)blockIdx.x * D_;
    float v1 = x[base + tid];
    float v2 = x[base + 256 + tid];
    red[tid] = v1 + v2;
    __syncthreads();
    for (int s = 128; s > 0; s >>= 1) { if (tid < s) red[tid] += red[tid + s]; __syncthreads(); }
    float mu = red[0] * (1.0f / D_);
    __syncthreads();
    float d1 = v1 - mu, d2 = v2 - mu;
    red[tid] = d1 * d1 + d2 * d2;
    __syncthreads();
    for (int s = 128; s > 0; s >>= 1) { if (tid < s) red[tid] += red[tid + s]; __syncthreads(); }
    float rstd = rsqrtf(red[0] * (1.0f / D_) + 1e-5f);
    xh[base + tid]       = (_Float16)(d1 * rstd * g[tid] + bta[tid]);
    xh[base + 256 + tid] = (_Float16)(d2 * rstd * g[256 + tid] + bta[256 + tid]);
}

// ---------------- Generic batched WMMA GEMM ----------------
// C[m][n] = sum_k A[m][k] * B[k][n] (+bias[n]),  with B[k][n] = Bw[n*ldb + k]
// Block: 128 threads = 4 waves; block tile 64x64; each wave 32x32 (2x2 frags).
// Requires M%64==0, N%64==0, K%32==0.  LDS stride 40 halfs keeps every
// fragment read and staging write 16B aligned (b128 LDS traffic).
template<bool A_F16, bool C_F16>
__global__ void k_wmma_gemm(const void* __restrict__ Ain,
                            const _Float16* __restrict__ Bw,
                            const float* __restrict__ bias,
                            void* __restrict__ Cout,
                            int K, int lda, int ldb, int ldc,
                            long aStride, long bStride,
                            long cStrideOuter, long cStrideInner, int cDiv) {
    __shared__ _Float16 As[64][40];
    __shared__ _Float16 Bs[64][40];

    const int tid  = threadIdx.x;        // 0..127
    const int lane = tid & 31;
    const int wave = tid >> 5;           // 0..3
    const int wm = (wave >> 1) * 32;
    const int wn = (wave & 1) * 32;
    const int m0 = blockIdx.y * 64;
    const int n0 = blockIdx.x * 64;
    const int z  = blockIdx.z;

    const _Float16* A16 = (const _Float16*)Ain + (size_t)z * aStride;
    const float*    A32 = (const float*)Ain    + (size_t)z * aStride;
    const _Float16* Bp  = Bw + (size_t)z * bStride;
    const size_t cOff = (size_t)(z / cDiv) * cStrideOuter + (size_t)(z % cDiv) * cStrideInner;

    const int sr = tid >> 1;             // staging row 0..63
    const int sk = (tid & 1) * 16;       // staging k-offset 0/16

    v8f c00 = {}, c01 = {}, c10 = {}, c11 = {};

    const int ml = lane & 15;
    const int kA = (lane < 16) ? 0 : 8;    // A: K base per lane-half (pairs)
    const int nl = lane & 15;
    const int kB = (lane < 16) ? 0 : 8;    // B: K base per lane-half

    for (int k0 = 0; k0 < K; k0 += 32) {
        // ---- stage A tile (64 x 32) ----
        if constexpr (A_F16) {
            const _Float16* ag = A16 + (size_t)(m0 + sr) * lda + k0 + sk;
            _Float16* ad = &As[sr][sk];
            copy16_g2lds(ad, ag);
            copy16_g2lds(ad + 8, ag + 8);
        } else {
            const float* ag = A32 + (size_t)(m0 + sr) * lda + k0 + sk;
            if (k0 + 32 < K) __builtin_prefetch(ag + 32, 0, 1);   // global_prefetch_b8
            union { v8h v; _Float16 e[8]; } t0, t1;
            #pragma unroll
            for (int i = 0; i < 8; ++i) {
                t0.e[i] = (_Float16)ag[i];
                t1.e[i] = (_Float16)ag[8 + i];
            }
            _Float16* ad = &As[sr][sk];
            *(v8h*)ad = t0.v;
            *(v8h*)(ad + 8) = t1.v;
        }
        // ---- stage B tile: Bs[n][k] = Bw[(n0+n)*ldb + k0+k] ----
        {
            const _Float16* bg = Bp + (size_t)(n0 + sr) * ldb + k0 + sk;
            _Float16* bd = &Bs[sr][sk];
            copy16_g2lds(bd, bg);
            copy16_g2lds(bd + 8, bg + 8);
        }
        async_join();
        __syncthreads();

        // ---- fragments per ISA 16x16x32 f16 layouts: all b128 LDS reads ----
        frag16 a0, a1, b0, b1;
        a0.h[0] = *(const v8h*)&As[wm + ml][kA];
        a0.h[1] = *(const v8h*)&As[wm + ml][16 + kA];
        a1.h[0] = *(const v8h*)&As[wm + 16 + ml][kA];
        a1.h[1] = *(const v8h*)&As[wm + 16 + ml][16 + kA];
        b0.h[0] = *(const v8h*)&Bs[wn + nl][kB];
        b0.h[1] = *(const v8h*)&Bs[wn + nl][16 + kB];
        b1.h[0] = *(const v8h*)&Bs[wn + 16 + nl][kB];
        b1.h[1] = *(const v8h*)&Bs[wn + 16 + nl][16 + kB];

        c00 = wmma_f16(a0.v, b0.v, c00);
        c01 = wmma_f16(a0.v, b1.v, c01);
        c10 = wmma_f16(a1.v, b0.v, c10);
        c11 = wmma_f16(a1.v, b1.v, c11);
        __syncthreads();
    }

    // ---- epilogue: C/D layout (VGPR r -> M=r / M=r+8 across lane halves) ----
    const int nc0 = n0 + wn + nl;
    const int nc1 = nc0 + 16;
    const float bv0 = bias ? bias[nc0] : 0.0f;
    const float bv1 = bias ? bias[nc1] : 0.0f;
    const int rb = wm + ((lane < 16) ? 0 : 8);
    #pragma unroll
    for (int r = 0; r < 8; ++r) {
        const size_t mr0 = (size_t)(m0 + rb + r) * ldc;
        const size_t mr1 = (size_t)(m0 + rb + 16 + r) * ldc;
        float v00 = c00[r] + bv0, v01 = c01[r] + bv1;
        float v10 = c10[r] + bv0, v11 = c11[r] + bv1;
        if constexpr (C_F16) {
            _Float16* Cp = (_Float16*)Cout + cOff;
            Cp[mr0 + nc0] = (_Float16)v00; Cp[mr0 + nc1] = (_Float16)v01;
            Cp[mr1 + nc0] = (_Float16)v10; Cp[mr1 + nc1] = (_Float16)v11;
        } else {
            float* Cp = (float*)Cout + cOff;
            Cp[mr0 + nc0] = v00; Cp[mr0 + nc1] = v01;
            Cp[mr1 + nc0] = v10; Cp[mr1 + nc1] = v11;
        }
    }
}

// ---------------- split heads, add pos biases, transpose V ----------------
__global__ void k_split_heads(const _Float16* __restrict__ qf, const _Float16* __restrict__ kf,
                              const _Float16* __restrict__ vf,
                              const float* __restrict__ pbu, const float* __restrict__ pbv,
                              _Float16* __restrict__ qu, _Float16* __restrict__ qv,
                              _Float16* __restrict__ kh, _Float16* __restrict__ vt) {
    int idx = blockIdx.x * blockDim.x + threadIdx.x;
    if (idx >= B_ * L_ * D_) return;
    int col = idx & (D_ - 1);
    int row = idx >> 9;
    int b = row / L_, l = row % L_;
    int h = col >> 6, d = col & 63;
    float q = (float)qf[idx];
    size_t hb = (((size_t)b * H_ + h) * L_ + l) * HD_ + d;
    qu[hb] = (_Float16)(q + pbu[col]);
    qv[hb] = (_Float16)(q + pbv[col]);
    kh[hb] = kf[idx];
    vt[(((size_t)b * H_ + h) * HD_ + d) * L_ + l] = vf[idx];
}

// ---------------- scores_bd: s[b,h,i,j] += sum_d qv[b,j,h,d]*pos[h,i,j,d] ----------------
// One wave per (h, j, 16-wide i-tile).  A = qv[:, j, h, :] (8 rows, padded), B = pos[h, itile, j, :]^T.
__global__ void k_scores_bd(const _Float16* __restrict__ qvh, const _Float16* __restrict__ posh,
                            float* __restrict__ scores) {
    const int lane = threadIdx.x & 31;
    const int w = blockIdx.x * 8 + (threadIdx.x >> 5);
    const int IT = L_ / 16;                 // 24
    const int it = w % IT;
    const int j  = (w / IT) % L_;
    const int h  = w / (IT * L_);

    const int ml = lane & 15;
    const int kA = (lane < 16) ? 0 : 8;
    const int nl = lane & 15;
    const int kB = (lane < 16) ? 0 : 8;

    v8f c = {};
    #pragma unroll
    for (int ks = 0; ks < 2; ++ks) {        // K = 64 = 2 x 32
        frag16 a; a.v = (v16h){};
        if (ml < B_) {
            const size_t ab = (((size_t)ml * H_ + h) * L_ + j) * HD_ + ks * 32;
            a.h[0] = *(const v8h*)&qvh[ab + kA];
            a.h[1] = *(const v8h*)&qvh[ab + 16 + kA];
        }
        frag16 bf;
        const int ig = it * 16 + nl;
        const size_t bb = ((size_t)ig * L_ + j) * D_ + h * HD_ + ks * 32 + kB;
        bf.h[0] = *(const v8h*)&posh[bb];
        bf.h[1] = *(const v8h*)&posh[bb + 16];
        c = wmma_f16(a.v, bf.v, c);
    }
    if (lane < 16) {                        // rows m=0..7 (= batch b) live in lanes 0-15
        const int ig = it * 16 + lane;
        #pragma unroll
        for (int r = 0; r < B_; ++r)
            scores[(((size_t)r * H_ + h) * L_ + ig) * L_ + j] += c[r];
    }
}

// ---------------- masked, scaled softmax over j; output f16 probs ----------------
__global__ void k_softmax(const float* __restrict__ scores, const int* __restrict__ mask,
                          _Float16* __restrict__ probs) {
    __shared__ float red[128];
    const int tid = threadIdx.x;
    const int row = blockIdx.x;             // (b*H + h)*L + i
    const int b = row / (H_ * L_);
    const size_t base = (size_t)row * L_;
    float v[3];
    #pragma unroll
    for (int t = 0; t < 3; ++t) {
        int j = tid + t * 128;
        float s = scores[base + j] * 0.125f; // 1/sqrt(64)
        if (mask[b * L_ + j] == 0) s = -3.0e38f;
        v[t] = s;
    }
    float m = fmaxf(v[0], fmaxf(v[1], v[2]));
    red[tid] = m; __syncthreads();
    for (int s = 64; s > 0; s >>= 1) { if (tid < s) red[tid] = fmaxf(red[tid], red[tid + s]); __syncthreads(); }
    m = red[0]; __syncthreads();
    float e[3], sum = 0.0f;
    #pragma unroll
    for (int t = 0; t < 3; ++t) { e[t] = __expf(v[t] - m); sum += e[t]; }
    red[tid] = sum; __syncthreads();
    for (int s = 64; s > 0; s >>= 1) { if (tid < s) red[tid] += red[tid + s]; __syncthreads(); }
    float inv = 1.0f / red[0];
    #pragma unroll
    for (int t = 0; t < 3; ++t) probs[base + tid + t * 128] = (_Float16)(e[t] * inv);
}

extern "C" void kernel_launch(void* const* d_in, const int* in_sizes, int n_in,
                              void* d_out, int out_size, void* d_ws, size_t ws_size,
                              hipStream_t stream) {
    (void)in_sizes; (void)n_in; (void)out_size; (void)ws_size;
    const float* hs    = (const float*)d_in[0];
    const float* pe    = (const float*)d_in[1];
    const int*   amask = (const int*)d_in[2];
    const float* ln_g  = (const float*)d_in[3];
    const float* ln_b  = (const float*)d_in[4];
    const float* Wq = (const float*)d_in[5];   const float* bq = (const float*)d_in[6];
    const float* Wk = (const float*)d_in[7];   const float* bk = (const float*)d_in[8];
    const float* Wv = (const float*)d_in[9];   const float* bv = (const float*)d_in[10];
    const float* Wpos = (const float*)d_in[11];
    const float* pbu  = (const float*)d_in[12];
    const float* pbv  = (const float*)d_in[13];
    const float* Wo = (const float*)d_in[14];  const float* bo = (const float*)d_in[15];

    char* w = (char*)d_ws;
    size_t off = 0;
    auto alloc = [&](size_t bytes) -> void* {
        void* p = w + off;
        off = (off + bytes + 255) & ~(size_t)255;
        return p;
    };
    const size_t DD = (size_t)D_ * D_;
    const size_t BLD = (size_t)B_ * L_ * D_;
    const size_t LLD = (size_t)L_ * L_ * D_;
    const size_t BHLL = (size_t)B_ * H_ * L_ * L_;

    _Float16* Wq_h   = (_Float16*)alloc(DD * 2);
    _Float16* Wk_h   = (_Float16*)alloc(DD * 2);
    _Float16* Wv_h   = (_Float16*)alloc(DD * 2);
    _Float16* Wpos_h = (_Float16*)alloc(DD * 2);
    _Float16* Wo_h   = (_Float16*)alloc(DD * 2);
    _Float16* xh     = (_Float16*)alloc(BLD * 2);
    _Float16* qf     = (_Float16*)alloc(BLD * 2);
    _Float16* kf     = (_Float16*)alloc(BLD * 2);
    _Float16* vf     = (_Float16*)alloc(BLD * 2);
    _Float16* qu     = (_Float16*)alloc(BLD * 2);
    _Float16* qv     = (_Float16*)alloc(BLD * 2);
    _Float16* kh     = (_Float16*)alloc(BLD * 2);
    _Float16* vt     = (_Float16*)alloc(BLD * 2);
    _Float16* ctx    = (_Float16*)alloc(BLD * 2);
    _Float16* posh   = (_Float16*)alloc(LLD * 2);
    float*    scores = (float*)alloc(BHLL * 4);
    _Float16* probs  = (_Float16*)alloc(BHLL * 2);

    // 1) weights -> f16
    {
        dim3 g((unsigned)((DD + 255) / 256)), b(256);
        k_f32_to_f16<<<g, b, 0, stream>>>(Wq,   Wq_h,   (int)DD);
        k_f32_to_f16<<<g, b, 0, stream>>>(Wk,   Wk_h,   (int)DD);
        k_f32_to_f16<<<g, b, 0, stream>>>(Wv,   Wv_h,   (int)DD);
        k_f32_to_f16<<<g, b, 0, stream>>>(Wpos, Wpos_h, (int)DD);
        k_f32_to_f16<<<g, b, 0, stream>>>(Wo,   Wo_h,   (int)DD);
    }
    // 2) LayerNorm + cast
    k_layernorm<<<dim3(B_ * L_), dim3(256), 0, stream>>>(hs, ln_g, ln_b, xh);

    // 3) q/k/v projections:  [3072,512] = xh @ W^T + b   (f16 out)
    {
        dim3 g(D_ / 64, (B_ * L_) / 64, 1), b(128);
        k_wmma_gemm<true,  true><<<g, b, 0, stream>>>(xh, Wq_h, bq, qf, D_, D_, D_, D_, 0, 0, 0, 0, 1);
        k_wmma_gemm<true,  true><<<g, b, 0, stream>>>(xh, Wk_h, bk, kf, D_, D_, D_, D_, 0, 0, 0, 0, 1);
        k_wmma_gemm<true,  true><<<g, b, 0, stream>>>(xh, Wv_h, bv, vf, D_, D_, D_, D_, 0, 0, 0, 0, 1);
    }
    // 4) split heads (+pos biases, V transpose)
    k_split_heads<<<dim3((B_ * L_ * D_) / 256), dim3(256), 0, stream>>>(
        qf, kf, vf, pbu, pbv, qu, qv, kh, vt);

    // 5) pos = position_embeddings @ Wpos^T : [147456,512]x[512,512], f32 A converted on stage-in
    k_wmma_gemm<false, true><<<dim3(D_ / 64, (L_ * L_) / 64, 1), dim3(128), 0, stream>>>(
        pe, Wpos_h, nullptr, posh, D_, D_, D_, D_, 0, 0, 0, 0, 1);

    // 6) scores_ac: per (b,h): Qu(384x64) @ K^T(64x384) -> f32 scores
    k_wmma_gemm<true, false><<<dim3(L_ / 64, L_ / 64, B_ * H_), dim3(128), 0, stream>>>(
        qu, kh, nullptr, scores, HD_, HD_, HD_, L_,
        (long)L_ * HD_, (long)L_ * HD_, (long)L_ * L_, 0, 1);

    // 7) scores_bd accumulate
    k_scores_bd<<<dim3((H_ * L_ * (L_ / 16)) / 8), dim3(256), 0, stream>>>(qv, posh, scores);

    // 8) softmax -> f16 probs
    k_softmax<<<dim3(B_ * H_ * L_), dim3(128), 0, stream>>>(scores, amask, probs);

    // 9) ctx: per (b,h): probs(384x384) @ V(384x64) -> ctx[b,i,h,d] scattered into [B*L, D]
    k_wmma_gemm<true, true><<<dim3(HD_ / 64, L_ / 64, B_ * H_), dim3(128), 0, stream>>>(
        probs, vt, nullptr, ctx, L_, L_, L_, D_,
        (long)L_ * L_, (long)HD_ * L_, (long)L_ * D_, (long)HD_, H_);

    // 10) out = ctx @ Wo^T + bo  (f32 out)
    k_wmma_gemm<true, false><<<dim3(D_ / 64, (B_ * L_) / 64, 1), dim3(128), 0, stream>>>(
        ctx, Wo_h, bo, (float*)d_out, D_, D_, D_, D_, 0, 0, 0, 0, 1);
}
